// RefAttnProcessor_15324443312825
// MI455X (gfx1250) — matrix operand order repaired and checked by reading, not compiled, not run on hardware
//
#include <hip/hip_runtime.h>

// ---------------------------------------------------------------------------
// Problem dims (fixed by the reference)
// ---------------------------------------------------------------------------
#define LQ 4096
#define LK 1024
#define NH 24
#define HD 128
#define EM 3072   // hidden = NH * HD

// LDS strides (bf16 elements) — padded to dodge bank conflicts, 16B aligned rows
#define KSTR  136   // 32 rows x 128 d (+8 pad) — TDM pad fields reproduce this
#define VTSTR 40    // 128 rows x 32 keys (+8 pad)
#define PSTR  40    // per-wave 16 rows x 32 keys (+8 pad)

typedef __attribute__((ext_vector_type(16))) __bf16 v16bf;
typedef __attribute__((ext_vector_type(8)))  __bf16 v8bf;
typedef __attribute__((ext_vector_type(8)))  float  v8f;
typedef __attribute__((ext_vector_type(4)))  unsigned int v4u;
typedef __attribute__((ext_vector_type(8)))  unsigned int v8u;

static __device__ __forceinline__ v8f wmma_bf16(v16bf a, v16bf b, v8f c) {
  // D = A(16x32 bf16) * B(32x16 bf16) + C(16x16 f32)
  return __builtin_amdgcn_wmma_f32_16x16x32_bf16(
      /*neg_a=*/false, a, /*neg_b=*/false, b,
      /*c_mod=*/(short)0, c, /*reuse_a=*/false, /*reuse_b=*/false);
}

static __device__ __forceinline__ v8f vzero8() {
  v8f z;
#pragma unroll
  for (int i = 0; i < 8; ++i) z[i] = 0.0f;
  return z;
}

static __device__ __forceinline__ v16bf comb(v8bf a, v8bf b) {
  return __builtin_shufflevector(a, b, 0, 1, 2, 3, 4, 5, 6, 7,
                                       8, 9, 10, 11, 12, 13, 14, 15);
}

// B-fragment: 16 contiguous f32 at p -> bf16 elements 0..15
static __device__ __forceinline__ v16bf load_cvt_b(const float* __restrict__ p) {
  const float4* q4 = (const float4*)p;
  float4 a = q4[0], b = q4[1], c = q4[2], d = q4[3];
  v16bf r;
  r[0]=(__bf16)a.x;  r[1]=(__bf16)a.y;  r[2]=(__bf16)a.z;  r[3]=(__bf16)a.w;
  r[4]=(__bf16)b.x;  r[5]=(__bf16)b.y;  r[6]=(__bf16)b.z;  r[7]=(__bf16)b.w;
  r[8]=(__bf16)c.x;  r[9]=(__bf16)c.y;  r[10]=(__bf16)c.z; r[11]=(__bf16)c.w;
  r[12]=(__bf16)d.x; r[13]=(__bf16)d.y; r[14]=(__bf16)d.z; r[15]=(__bf16)d.w;
  return r;
}

// A-fragment: elements 0..15 = p[0..15] (f32 -> bf16)
static __device__ __forceinline__ v16bf load_cvt_a(const float* __restrict__ p) {
  const float4* q4 = (const float4*)p;
  float4 a = q4[0], b = q4[1], c = q4[2], d = q4[3];
  v16bf r;
  r[0]=(__bf16)a.x;  r[1]=(__bf16)a.y;  r[2]=(__bf16)a.z;  r[3]=(__bf16)a.w;
  r[4]=(__bf16)b.x;  r[5]=(__bf16)b.y;  r[6]=(__bf16)b.z;  r[7]=(__bf16)b.w;
  r[8]=(__bf16)c.x;  r[9]=(__bf16)c.y;  r[10]=(__bf16)c.z; r[11]=(__bf16)c.w;
  r[12]=(__bf16)d.x; r[13]=(__bf16)d.y; r[14]=(__bf16)d.z; r[15]=(__bf16)d.w;
  return r;
}

static __device__ __forceinline__ v16bf load_cvt_a_scale(const float* __restrict__ p,
                                                         float s) {
  const float4* q4 = (const float4*)p;
  float4 a = q4[0], b = q4[1], c = q4[2], d = q4[3];
  v16bf r;
  r[0]=(__bf16)(a.x*s);  r[1]=(__bf16)(a.y*s);  r[2]=(__bf16)(a.z*s);  r[3]=(__bf16)(a.w*s);
  r[4]=(__bf16)(b.x*s);  r[5]=(__bf16)(b.y*s);  r[6]=(__bf16)(b.z*s);  r[7]=(__bf16)(b.w*s);
  r[8]=(__bf16)(c.x*s);  r[9]=(__bf16)(c.y*s);  r[10]=(__bf16)(c.z*s); r[11]=(__bf16)(c.w*s);
  r[12]=(__bf16)(d.x*s); r[13]=(__bf16)(d.y*s); r[14]=(__bf16)(d.z*s); r[15]=(__bf16)(d.w*s);
  return r;
}

// ---------------------------------------------------------------------------
// Kernel 1: K/V projection GEMM. out[l,n] = sum_e lat[l,e] * W[n,e]
//   block = 8 waves -> 32(M) x 256(N) tile; each wave 16x64 (4 accumulators)
//   z==0 -> K written as f32 (needs RMSNorm+RoPE after), z==1 -> V as bf16
// ---------------------------------------------------------------------------
__global__ __launch_bounds__(256) void kvproj_kernel(
    const float* __restrict__ lat, const float* __restrict__ Wk,
    const float* __restrict__ Wv, float* __restrict__ kout,
    __bf16* __restrict__ vout) {
  const int z = blockIdx.z;
  const float* __restrict__ W = z ? Wv : Wk;
  const int tid  = threadIdx.x;
  const int wave = tid >> 5;
  const int lane = tid & 31;
  const int l16  = lane & 15;
  const int lhalf = lane >> 4;
  const int mRow = blockIdx.x * 32 + (wave >> 2) * 16 + l16;   // A row for this lane
  const int nCol = blockIdx.y * 256 + (wave & 3) * 64;         // wave's N base

  v8f acc[4];
#pragma unroll
  for (int j = 0; j < 4; ++j) acc[j] = vzero8();

  const float* __restrict__ ap0 = lat + (size_t)mRow * EM;
  for (int e0 = 0; e0 < EM; e0 += 32) {
    v16bf af = load_cvt_a(ap0 + e0 + lhalf * 16);
#pragma unroll
    for (int j = 0; j < 4; ++j) {
      const float* bp = W + (size_t)(nCol + j * 16 + l16) * EM + e0 + lhalf * 16;
      __builtin_prefetch(bp + 32, 0, 1);     // global_prefetch next e-chunk
      v16bf bfr = load_cvt_b(bp);
      acc[j] = wmma_bf16(af, bfr, acc[j]);
    }
  }

#pragma unroll
  for (int j = 0; j < 4; ++j) {
#pragma unroll
    for (int v = 0; v < 8; ++v) {
      const int r = blockIdx.x * 32 + (wave >> 2) * 16 + v + 8 * lhalf;
      const int c = nCol + j * 16 + l16;
      if (z == 0) kout[(size_t)r * EM + c] = acc[j][v];
      else        vout[(size_t)r * EM + c] = (__bf16)acc[j][v];
    }
  }
}

// ---------------------------------------------------------------------------
// Kernel 2: RMSNorm + RoPE on K heads; f32 in, bf16 out.
//   One 64-thread block per (l, h); thread i owns the interleaved pair (2i,2i+1).
// ---------------------------------------------------------------------------
__global__ __launch_bounds__(64) void knorm_rope_kernel(
    const float* __restrict__ kf32, const float* __restrict__ freqs,
    const float* __restrict__ rmsw, __bf16* __restrict__ kbf) {
  const int blk = blockIdx.x;          // l * NH + h
  const int l   = blk / NH;
  const int i   = threadIdx.x;         // 0..63
  const size_t base = (size_t)blk * HD;

  const float x0 = kf32[base + 2 * i];
  const float x1 = kf32[base + 2 * i + 1];
  float ss = x0 * x0 + x1 * x1;
#pragma unroll
  for (int mask = 1; mask < 32; mask <<= 1) ss += __shfl_xor(ss, mask, 32);

  __shared__ float red[2];
  if ((i & 31) == 0) red[i >> 5] = ss;
  __syncthreads();
  const float tot = red[0] + red[1];
  const float rstd = rsqrtf(tot * (1.0f / (float)HD) + 1e-6f);

  const float n0 = x0 * rstd * rmsw[2 * i];
  const float n1 = x1 * rstd * rmsw[2 * i + 1];
  const float f  = freqs[(size_t)l * (HD / 2) + i];
  float sv, cv;
  __sincosf(f, &sv, &cv);
  kbf[base + 2 * i]     = (__bf16)(n0 * cv - n1 * sv);
  kbf[base + 2 * i + 1] = (__bf16)(n0 * sv + n1 * cv);
}

// ---------------------------------------------------------------------------
// Kernel 3: flash attention. block = 8 waves, one head, 128 q-rows.
//   K block staged global->LDS by the Tensor Data Mover (with descriptor
//   padding giving the 272B LDS row stride); V staged transposed by threads
//   in parallel with the TDM copy.
// ---------------------------------------------------------------------------
__global__ __launch_bounds__(256) void attn_kernel(
    const float* __restrict__ q, const __bf16* __restrict__ kbf,
    const __bf16* __restrict__ vbf, float* __restrict__ out) {
  __shared__ __bf16 Klds[32 * KSTR];        // [key(32)][d(128)+pad]
  __shared__ __bf16 VTlds[HD * VTSTR];      // [d(128)][key(32)+pad]
  __shared__ __bf16 Plds[8 * 16 * PSTR];    // per-wave [row(16)][key(32)+pad]

  const int h    = blockIdx.y;
  const int tid  = threadIdx.x;
  const int wave = tid >> 5;
  const int lane = tid & 31;
  const int l16  = lane & 15;
  const int lhalf = lane >> 4;

  // Q tile (16 rows x 128 d) as 4 A-fragments, softmax scale folded in
  const int qRow = blockIdx.x * 128 + wave * 16 + l16;
  const float* __restrict__ qp = q + ((size_t)qRow * NH + h) * HD;
  const float sc = 0.08838834764831845f;    // 1/sqrt(128)
  v16bf qf[4];
#pragma unroll
  for (int kd = 0; kd < 4; ++kd)
    qf[kd] = load_cvt_a_scale(qp + kd * 32 + lhalf * 16, sc);

  v8f o[8];
  float m[8], l[8];
#pragma unroll
  for (int i = 0; i < 8; ++i) { o[i] = vzero8(); m[i] = -3.0e38f; l[i] = 0.0f; }

  const unsigned kldsAddr = (unsigned)(uintptr_t)&Klds[0];

  for (int kb = 0; kb < LK / 32; ++kb) {
    __syncthreads();   // all readers of the previous block are done

    // --- TDM: K block (32 rows x 256B, memory row stride 6144B) -> LDS ---
    if (wave == 0) {
      const uint64_t ga =
          (uint64_t)(uintptr_t)(kbf + (((size_t)(kb * 32)) * NH + h) * HD);
      v4u g0;
      g0[0] = 1u;                                     // count=1, user mode
      g0[1] = kldsAddr;                               // lds_addr
      g0[2] = (unsigned)(ga & 0xffffffffu);           // global_addr[31:0]
      g0[3] = (unsigned)(ga >> 32) | 0x80000000u;     // global_addr[56:32] | type=2
      v8u g1;
      // data_size=2B (code 1), pad_enable, pad_interval=64DW (code 5),
      // pad_amount=4DW (code 3)  => LDS row stride 256+16B = KSTR*2
      g1[0] = (1u << 16) | (1u << 20) | (5u << 22) | (3u << 25);
      g1[1] = (unsigned)HD << 16;                     // tensor_dim0 = 128
      g1[2] = 32u << 16;                              // tensor_dim1 = 32
      g1[3] = (unsigned)HD << 16;                     // tile_dim0 = 128
      g1[4] = 32u;                                    // tile_dim1 = 32, tile_dim2 = 0
      g1[5] = (unsigned)EM;                           // tensor_dim0_stride = 3072 elems
      g1[6] = 0u;
      g1[7] = 0u;
      asm volatile("tensor_load_to_lds %0, %1" :: "s"(g0), "s"(g1) : "memory");
    }

    // --- threads stage V transposed while the TDM copy is in flight ---
    {
      const int row = tid >> 3;   // key in block: 0..31
      const int seg = tid & 7;    // 16-element d segment
      const __bf16* vp =
          vbf + (((size_t)(kb * 32 + row)) * NH + h) * HD + seg * 16;
      v8bf v0 = *(const v8bf*)vp;
      v8bf v1 = *(const v8bf*)(vp + 8);
#pragma unroll
      for (int j = 0; j < 8; ++j) {
        VTlds[(seg * 16 + j) * VTSTR + row]     = v0[j];
        VTlds[(seg * 16 + 8 + j) * VTSTR + row] = v1[j];
      }
    }
    if (wave == 0) __builtin_amdgcn_s_wait_tensorcnt(0);
    __syncthreads();

    // S(16x32) = Q(16x128) @ K^T : two 16-key N-tiles, 4 k-dim steps each
    v8f s0 = vzero8(), s1 = vzero8();
#pragma unroll
    for (int kd = 0; kd < 4; ++kd) {
      const __bf16* b0p = &Klds[l16 * KSTR + kd * 32 + lhalf * 16];
      const __bf16* b1p = &Klds[(16 + l16) * KSTR + kd * 32 + lhalf * 16];
      v16bf b0 = comb(*(const v8bf*)b0p, *(const v8bf*)(b0p + 8));
      v16bf b1 = comb(*(const v8bf*)b1p, *(const v8bf*)(b1p + 8));
      s0 = wmma_bf16(qf[kd], b0, s0);
      s1 = wmma_bf16(qf[kd], b1, s1);
    }

    // Online softmax. C-layout: VGPR v, lane -> row (v + 8*lhalf), col lane%16.
#pragma unroll
    for (int v = 0; v < 8; ++v) {
      float t = fmaxf(s0[v], s1[v]);
      t = fmaxf(t, __shfl_xor(t, 1, 32));
      t = fmaxf(t, __shfl_xor(t, 2, 32));
      t = fmaxf(t, __shfl_xor(t, 4, 32));
      t = fmaxf(t, __shfl_xor(t, 8, 32));
      const float mn   = fmaxf(m[v], t);
      const float corr = __expf(m[v] - mn);
      m[v] = mn;
      const float p0 = __expf(s0[v] - mn);
      const float p1 = __expf(s1[v] - mn);
      float rs = p0 + p1;
      rs += __shfl_xor(rs, 1, 32);
      rs += __shfl_xor(rs, 2, 32);
      rs += __shfl_xor(rs, 4, 32);
      rs += __shfl_xor(rs, 8, 32);
      l[v] = l[v] * corr + rs;
#pragma unroll
      for (int dt = 0; dt < 8; ++dt) o[dt][v] *= corr;
      // bounce P through LDS to convert C-layout -> A-layout
      const int pr = (wave * 16 + v + 8 * lhalf) * PSTR + l16;
      Plds[pr]      = (__bf16)p0;
      Plds[pr + 16] = (__bf16)p1;
    }
    asm volatile("s_wait_dscnt 0x0" ::: "memory");  // per-wave LDS RAW fence

    const __bf16* pp = &Plds[(wave * 16 + l16) * PSTR + lhalf * 16];
    v16bf pf = comb(*(const v8bf*)pp, *(const v8bf*)(pp + 8));

    // O(16x128) += P(16x32) @ V(32x128) : 8 N-tiles from transposed-V LDS
#pragma unroll
    for (int dt = 0; dt < 8; ++dt) {
      const __bf16* vtp = &VTlds[(dt * 16 + l16) * VTSTR + lhalf * 16];
      v16bf vfrag = comb(*(const v8bf*)vtp, *(const v8bf*)(vtp + 8));
      o[dt] = wmma_bf16(pf, vfrag, o[dt]);
    }
  }

  // epilogue: normalize by row sums, write f32 output [Lq, H*D]
#pragma unroll
  for (int v = 0; v < 8; ++v) {
    const float inv = 1.0f / l[v];
    const int qr = blockIdx.x * 128 + wave * 16 + v + 8 * lhalf;
    float* op = out + (size_t)qr * EM + h * HD;
#pragma unroll
    for (int dt = 0; dt < 8; ++dt)
      op[dt * 16 + l16] = o[dt][v] * inv;
  }
}

// ---------------------------------------------------------------------------
// Launch
// ---------------------------------------------------------------------------
extern "C" void kernel_launch(void* const* d_in, const int* in_sizes, int n_in,
                              void* d_out, int out_size, void* d_ws, size_t ws_size,
                              hipStream_t stream) {
  (void)in_sizes; (void)n_in; (void)out_size; (void)ws_size;
  const float* query  = (const float*)d_in[0];   // [1,4096,24,128]
  const float* latent = (const float*)d_in[1];   // [1,1024,3072]
  const float* freqs  = (const float*)d_in[2];   // [1024,64]
  const float* Wk     = (const float*)d_in[3];   // [3072,3072]
  const float* Wv     = (const float*)d_in[4];   // [3072,3072]
  const float* rmsw   = (const float*)d_in[5];   // [128]

  char* ws = (char*)d_ws;
  float*  kf32 = (float*)(ws);                                  // 12,582,912 B
  __bf16* kbf  = (__bf16*)(ws + (size_t)LK * EM * 4);           //  6,291,456 B
  __bf16* vbf  = (__bf16*)(ws + (size_t)LK * EM * 4
                              + (size_t)LK * EM * 2);           //  6,291,456 B

  kvproj_kernel<<<dim3(LK / 32, EM / 256, 2), 256, 0, stream>>>(
      latent, Wk, Wv, kf32, vbf);
  knorm_rope_kernel<<<dim3(LK * NH), 64, 0, stream>>>(kf32, freqs, rmsw, kbf);
  attn_kernel<<<dim3(LQ / 128, NH), 256, 0, stream>>>(
      query, kbf, vbf, (float*)d_out);
}
